// Semantic_Mapping_7052336300215
// MI455X (gfx1250) — matrix operand order repaired and checked by reading, not compiled, not run on hardware
//
#include <hip/hip_runtime.h>

// ---------------- problem constants ----------------
#define B_      4
#define NSEM    12
#define H_      480
#define W_      640
#define HW      (H_ * W_)
#define NCH     16              // obs channels: 3 rgb + 1 depth + 12 sem
#define VR_     100
#define NZ_     80              // MAX_H - MIN_H = 72 - (-8)
#define NF_     13              // 1 occupancy + 12 sem
#define MINZ_   13              // 25/5 - (-8)
#define MAXZ_   25              // int(89/5) - (-8)
#define SHIFTX  250.0f          // VR*int(RES)//2
#define AGH     88.0f
#define XC_     319.5f
#define ZC_     239.5f
#define GRID_ELEMS ((long)B_ * VR_ * VR_ * NZ_ * NF_)   // 41,600,000 f32 = 166.4 MB (fits global L2)
#define PAD_ELEMS  1024         // zeroed tail: reduce reads z up to 95 past the last column

typedef __attribute__((ext_vector_type(16))) _Float16 v16h;
typedef __attribute__((ext_vector_type(8)))  float    v8f;

// ---------------- kernel 1: zero the voxel grid (+ padding tail) ------------
__global__ void k_zero(float4* __restrict__ ws, long n4) {
    long i = (long)blockIdx.x * blockDim.x + threadIdx.x;
    long stride = (long)gridDim.x * blockDim.x;
    float4 z = make_float4(0.f, 0.f, 0.f, 0.f);
    for (; i < n4; i += stride) ws[i] = z;
}

// ---------------- kernel 2: trilinear splat (scatter with L2 f32 atomics) ----
// grid layout: [b][X(100)][Y(100)][Z(80)][F(13)]  -> one corner touches 13
// consecutive floats (52B) => 1-2 cachelines per corner for atomic locality.
__global__ void k_scatter(const float* __restrict__ obs, float* __restrict__ grid) {
    int gid = blockIdx.x * blockDim.x + threadIdx.x;
    if (gid >= B_ * HW) return;
    int b   = gid / HW;
    int pix = gid - b * HW;
    int h   = pix / W_;
    int w   = pix - h * W_;

    const float f_cam = (W_ * 0.5f) / tanf(0.017453292519943295f * (79.0f * 0.5f));

    float d    = obs[((long)b * NCH + 3) * HW + pix];
    float x_pc = ((float)w - XC_) * d / f_cam;
    float z_pc = (ZC_ - (float)h) * d / f_cam;
    float pos_x = (x_pc + SHIFTX) * 0.2f;        // x_w / RES
    float pos_y = d * 0.2f;                      // y_w / RES
    float pos_z = (z_pc + AGH) * 0.2f + 8.0f;    // z_w / ZRES - MIN_H

    float sv[NSEM];
#pragma unroll
    for (int s = 0; s < NSEM; ++s)
        sv[s] = obs[((long)b * NCH + 4 + s) * HW + pix];

    int   ix = (int)floorf(pos_x); float fx = pos_x - (float)ix;
    int   iy = (int)floorf(pos_y); float fy = pos_y - (float)iy;
    int   iz = (int)floorf(pos_z); float fz = pos_z - (float)iz;

#pragma unroll
    for (int cx = 0; cx < 2; ++cx) {
        int X = ix + cx; float wx = cx ? fx : 1.f - fx;
        bool okx = (X > 0) && (X < VR_);
#pragma unroll
        for (int cy = 0; cy < 2; ++cy) {
            int Y = iy + cy; float wy = cy ? fy : 1.f - fy;
            bool oky = (Y > 0) && (Y < VR_);
#pragma unroll
            for (int cz = 0; cz < 2; ++cz) {
                int Z = iz + cz; float wz = cz ? fz : 1.f - fz;
                bool okz = (Z > 0) && (Z < NZ_);
                if (okx && oky && okz) {
                    float wgt = wx * wy * wz;
                    if (wgt > 0.f) {
                        float* p = grid + ((((long)b * VR_ + X) * VR_ + Y) * NZ_ + Z) * NF_;
                        unsafeAtomicAdd(p, wgt);               // occupancy ("ones")
#pragma unroll
                        for (int s = 0; s < NSEM; ++s)
                            if (sv[s] != 0.f)                  // skip zero features (~90%)
                                unsafeAtomicAdd(p + 1 + s, wgt * sv[s]);
                    }
                }
            }
        }
    }
}

// ---------------- kernel 3: round + dual z-reduction as WMMA mask-matmul ----
// One wave handles (b, x, 16 y-columns).  D = A(16x32 f16 mask) * B(32x16 f16
// rounded counts) accumulated f32 over 3 K-chunks (z=0..95; z>=80 killed by
// the A mask, so B loads are UNCONDITIONAL — the zeroed padding tail keeps the
// final overrun finite).
//   A row m=0 : all-height mask (z < 80)
//   A row m=1 : agent-band mask (13 <= z < 25)
// D[M=0,N] / D[M=1,N] land in accumulator VGPR 0 / 1, lanes 0-15 (ISA layout).
__global__ void __launch_bounds__(256) k_reduce(const float* __restrict__ grid,
                                                float* __restrict__ out) {
    const int NYT = 7;                               // ceil(100/16)
    int tid  = blockIdx.x * blockDim.x + threadIdx.x;
    int wid  = tid >> 5;
    int lane = threadIdx.x & 31;
    if (wid >= B_ * VR_ * NYT) return;

    int b    = wid / (VR_ * NYT);
    int rem  = wid - b * (VR_ * NYT);
    int x    = rem / NYT;
    int y0   = (rem - x * NYT) * 16;

    int  m     = lane & 15;                          // A: M-row  | B: N-col
    bool upper = lane >= 16;
    int  y     = y0 + m;
    bool yok   = y < VR_;
    int  ysafe = yok ? y : (VR_ - 1);                // clamped: loads always valid/finite
    const float* col = grid + (((long)b * VR_ + x) * VR_ + ysafe) * NZ_ * NF_;

    // Build the three A (mask) chunks once; f16 A-layout: lane(0-15)=M,
    // elems 0..15 -> K=e(+16 upper), matching the B fill.
    v16h A[3];
#pragma unroll
    for (int c = 0; c < 3; ++c) {
        v16h a;
#pragma unroll
        for (int e = 0; e < 16; ++e) {
            int K = e + (upper ? 16 : 0);
            int z = 32 * c + K;
            float v = 0.f;
            if (m == 0)      v = (z < NZ_) ? 1.f : 0.f;
            else if (m == 1) v = (z >= MINZ_ && z < MAXZ_) ? 1.f : 0.f;
            a[e] = (_Float16)v;
        }
        A[c] = a;
    }

    int kofs = upper ? 16 : 0;                       // B-layout: upper lanes carry K+16

#pragma unroll
    for (int f = 0; f < NF_; ++f) {
        v8f acc = {0.f, 0.f, 0.f, 0.f, 0.f, 0.f, 0.f, 0.f};
#pragma unroll
        for (int c = 0; c < 3; ++c) {
            const float* cp = col + f + (long)(32 * c + kofs) * NF_;
            float raw[16];
#pragma unroll
            for (int e = 0; e < 16; ++e)             // unconditional burst of 16 loads
                raw[e] = cp[e * NF_];                // constant byte offsets e*52
            v16h bm;
#pragma unroll
            for (int e = 0; e < 16; ++e)
                bm[e] = (_Float16)rintf(raw[e]);     // jnp.round (RNE), f16-exact counts
            acc = __builtin_amdgcn_wmma_f32_16x16x32_f16(
                      false, A[c], false, bm, (short)0, acc, false, false);
        }
        if (!upper && yok) {
            float allv = acc[0];                     // D[M=0, N=lane]
            float agv  = acc[1];                     // D[M=1, N=lane]
            int   sp   = y * VR_ + x;                // output is swapaxes(2,3): [b,f,y,x]
            if (f == 0) {
                out[(long)b * 10000 + sp]            = fminf(fmaxf(agv,  0.f), 1.f);  // fp_map
                out[(long)(B_ + b) * 10000 + sp]     = fminf(fmaxf(allv, 0.f), 1.f);  // fp_exp
            } else {
                out[(long)(2 * B_) * 10000 + ((long)b * NSEM + (f - 1)) * 10000 + sp]
                    = fminf(fmaxf(agv * 0.2f, 0.f), 1.f);                             // sem_map
            }
        }
    }
}

// ---------------- host ----------------
extern "C" void kernel_launch(void* const* d_in, const int* in_sizes, int n_in,
                              void* d_out, int out_size, void* d_ws, size_t ws_size,
                              hipStream_t stream) {
    const float* obs  = (const float*)d_in[0];
    float*       grid = (float*)d_ws;
    float*       out  = (float*)d_out;

    if (ws_size < (size_t)(GRID_ELEMS + PAD_ELEMS) * sizeof(float)) return;

    long n4 = (GRID_ELEMS + PAD_ELEMS) / 4;
    k_zero<<<dim3(8192), dim3(256), 0, stream>>>((float4*)grid, n4);

    int np = B_ * HW;
    k_scatter<<<dim3((np + 255) / 256), dim3(256), 0, stream>>>(obs, grid);

    // 4 * 100 * 7 = 2800 waves -> 89600 threads -> 350 blocks of 256
    k_reduce<<<dim3(350), dim3(256), 0, stream>>>(grid, out);
}
